// CLQDNN_6133213299315
// MI455X (gfx1250) — matrix operand order repaired and checked by reading, compiled, not run on hardware
//
#include <hip/hip_runtime.h>
#include <hip/hip_bf16.h>
#include <math.h>

// Problem constants (must match reference)
#define VOCAB 50000
#define DIM   128
#define UNITS 256
#define BATCH 32
#define SEQ   2048
#define M_TOK (BATCH*SEQ)      // 65536 tokens
#define KDIM  256              // [real|imag]
#define NDIM  512              // [A-part | B-part]

// LDS tile row stride in halves: 40 halves = 80 B = 20 dwords.
// 16-lane b128 fragment reads advance 20 dwords/row -> 16 distinct banks mod 64,
// conflict-free, and 80 is a multiple of 16 B so b128 alignment holds.
#define LDS_STRIDE 40

typedef __attribute__((ext_vector_type(16))) _Float16 v16h;
typedef __attribute__((ext_vector_type(8)))  float    v8f;

union FragU {
    v16h h;
    uint4 u[2];
};

// ---------------------------------------------------------------------------
// Kernel 1: build Wt (NDIM x KDIM, f16, row-major by output column u')
//   u' < 256 (A-part):  Wt[u',k] = rk[u',k]        (k<128),  -ik[u',k-128] (k>=128)
//   u' >=256 (B-part):  Wt[u',k] = ik[u'-256,k]    (k<128),   rk[u'-256,k-128]
// kernel input layout: (UNITS, D, 2) f32
// ---------------------------------------------------------------------------
__global__ void prep_w_kernel(const float* __restrict__ kern, _Float16* __restrict__ Wt)
{
    int u = blockIdx.x;          // 0..511
    int k = threadIdx.x;         // 0..255
    int uu = u & 255;
    bool isB = (u >= 256);
    float v;
    if (k < DIM) {
        float rk = kern[uu * (DIM * 2) + k * 2 + 0];
        float ik = kern[uu * (DIM * 2) + k * 2 + 1];
        v = isB ? ik : rk;
    } else {
        int d = k - DIM;
        float rk = kern[uu * (DIM * 2) + d * 2 + 0];
        float ik = kern[uu * (DIM * 2) + d * 2 + 1];
        v = isB ? rk : -ik;
    }
    Wt[(size_t)u * KDIM + k] = (_Float16)v;
}

// ---------------------------------------------------------------------------
// Kernel 2: per-token gather + normalize + phase rotation.
// One block (128 threads) per token. Coalesced row gather from both tables.
// ---------------------------------------------------------------------------
__global__ __launch_bounds__(128) void prep_tok_kernel(
    const int*   __restrict__ indices,
    const float* __restrict__ amp_table,
    const float* __restrict__ phase_table,
    _Float16*    __restrict__ X,
    float*       __restrict__ wraw)
{
    __shared__ float red[128];
    __shared__ float ss_sh;

    int m = blockIdx.x;          // token id
    int d = threadIdx.x;         // 0..127
    int idx = indices[m];

    float a = amp_table[(size_t)idx * DIM + d];
    float p = phase_table[(size_t)idx * DIM + d];

    red[d] = a * a;
    __syncthreads();
    for (int off = 64; off > 0; off >>= 1) {
        if (d < off) red[d] += red[d + off];
        __syncthreads();
    }
    if (d == 0) {
        float ss = red[0];
        ss_sh = ss;
        wraw[m] = sqrtf(1e-5f + ss);
    }
    __syncthreads();

    float ss   = ss_sh;
    float inv  = 1.0f / fmaxf(sqrtf(ss), 1e-12f);
    float an   = a * inv;
    float re   = __cosf(p) * an;
    float im   = __sinf(p) * an;

    _Float16* xrow = X + (size_t)m * KDIM;
    xrow[d]       = (_Float16)re;
    xrow[DIM + d] = (_Float16)im;
}

// ---------------------------------------------------------------------------
// Kernel 3: softmax over S per batch row. One block (256 threads) per b.
// ---------------------------------------------------------------------------
__global__ __launch_bounds__(256) void softmax_kernel(
    const float* __restrict__ wraw, float* __restrict__ wout)
{
    __shared__ float red[256];
    __shared__ float stat;

    int b   = blockIdx.x;
    int tid = threadIdx.x;
    const float* row = wraw + (size_t)b * SEQ;

    float mx = -3.0e38f;
    for (int s = tid; s < SEQ; s += 256) mx = fmaxf(mx, row[s]);
    red[tid] = mx;
    __syncthreads();
    for (int off = 128; off > 0; off >>= 1) {
        if (tid < off) red[tid] = fmaxf(red[tid], red[tid + off]);
        __syncthreads();
    }
    if (tid == 0) stat = red[0];
    __syncthreads();
    float bmax = stat;

    float sum = 0.0f;
    for (int s = tid; s < SEQ; s += 256) sum += __expf(row[s] - bmax);
    __syncthreads();
    red[tid] = sum;
    __syncthreads();
    for (int off = 128; off > 0; off >>= 1) {
        if (tid < off) red[tid] += red[tid + off];
        __syncthreads();
    }
    if (tid == 0) stat = red[0];
    __syncthreads();
    float inv = 1.0f / stat;

    float* orow = wout + (size_t)b * SEQ;
    for (int s = tid; s < SEQ; s += 256) orow[s] = __expf(row[s] - bmax) * inv;
}

// ---------------------------------------------------------------------------
// Kernel 4: zero the output (atomic accumulation target)
// ---------------------------------------------------------------------------
__global__ void zero_out_kernel(float* __restrict__ out, int n)
{
    int i = blockIdx.x * blockDim.x + threadIdx.x;
    if (i < n) out[i] = 0.0f;
}

// ---------------------------------------------------------------------------
// Kernel 5: LDS-staged WMMA GEMM  Y = X(65536x256) @ Wt^T(256x512)
// with fused epilogue  out[b, u mod 256] += w[m] * Y[m,u]^2
//
// Block = 256 threads = 8 waves, output tile 128(M) x 128(N).
// Wave grid 4(M) x 2(N); each wave: 32 rows x 64 cols
//   = 8 accumulators of v_wmma_f32_16x16x32_f16, K-loop 256 in steps of 32.
// Per K-step both the 128x32 A-tile and 128x32 B-tile are staged in LDS
// (padded stride, bank-conflict-free b128 fragment reads), cutting global/L2
// traffic ~3x vs per-wave global fragment loads.
//
// Fragment layouts per CDNA5 ISA 7.12.2 (wave32):
//   A 16x32 f16: lane l holds row M=l&15, K chunks {sel*8..+7, 16+sel*8..+7}
//   B 32x16 f16: lane l holds col N=l&15, K chunks {sel*8..+7, 16+sel*8..+7}
//   C 16x16 f32: lane l col N=l&15; VGPR r holds row M = r + 8*sel
// ---------------------------------------------------------------------------
__global__ __launch_bounds__(256) void gemm_reduce_kernel(
    const _Float16* __restrict__ X,
    const _Float16* __restrict__ Wt,
    const float*    __restrict__ w,
    float*          __restrict__ out)
{
    __shared__ _Float16 smA[128 * LDS_STRIDE];   // 10 KB
    __shared__ _Float16 smB[128 * LDS_STRIDE];   // 10 KB
    __shared__ float    smOut[128];

    const int tid  = threadIdx.x;
    const int lane = tid & 31;
    const int wave = tid >> 5;
    const int sel  = lane >> 4;       // 0/1: upper half-wave
    const int l16  = lane & 15;

    const int m0 = blockIdx.x * 128;              // block row base
    const int n0 = blockIdx.y * 128;              // block col base
    const int mw = 32 * (wave & 3);               // wave row offset in tile
    const int nw = 64 * (wave >> 2);              // wave col offset in tile

    // staging assignment: 2 threads per row/col, 16 halves (32 B) each
    const int srow  = tid >> 1;                   // 0..127
    const int schnk = (tid & 1) * 16;             // half offset 0 or 16

    v8f c[2][4];
#pragma unroll
    for (int i = 0; i < 2; ++i)
#pragma unroll
        for (int j = 0; j < 4; ++j)
            c[i][j] = (v8f){0.f,0.f,0.f,0.f,0.f,0.f,0.f,0.f};

    const _Float16* gA = X  + (size_t)(m0 + srow) * KDIM + schnk;
    const _Float16* gB = Wt + (size_t)(n0 + srow) * KDIM + schnk;
    _Float16* sA = smA + srow * LDS_STRIDE + schnk;
    _Float16* sB = smB + srow * LDS_STRIDE + schnk;

#pragma unroll
    for (int k = 0; k < KDIM; k += 32) {
        __syncthreads();                           // protect previous stage
        // stage A/B K-slices into LDS (32 B per thread per matrix)
        *(uint4*)(sA)     = *(const uint4*)(gA + k);
        *(uint4*)(sA + 8) = *(const uint4*)(gA + k + 8);
        *(uint4*)(sB)     = *(const uint4*)(gB + k);
        *(uint4*)(sB + 8) = *(const uint4*)(gB + k + 8);
        __syncthreads();

        FragU a0, a1;
        // A fragment: row = mw + l16 (+16), K chunks {sel*8, 16+sel*8}
        const _Float16* ar0 = smA + (mw + l16)      * LDS_STRIDE;
        const _Float16* ar1 = smA + (mw + 16 + l16) * LDS_STRIDE;
        a0.u[0] = *(const uint4*)(ar0 + sel * 8);
        a0.u[1] = *(const uint4*)(ar0 + 16 + sel * 8);
        a1.u[0] = *(const uint4*)(ar1 + sel * 8);
        a1.u[1] = *(const uint4*)(ar1 + 16 + sel * 8);
#pragma unroll
        for (int j = 0; j < 4; ++j) {
            FragU bf;
            // B fragment: col = nw + 16j + l16, K chunks {sel*8, 16+sel*8}
            const _Float16* bc = smB + (nw + 16 * j + l16) * LDS_STRIDE;
            bf.u[0] = *(const uint4*)(bc + sel * 8);
            bf.u[1] = *(const uint4*)(bc + 16 + sel * 8);
            c[0][j] = __builtin_amdgcn_wmma_f32_16x16x32_f16(
                false, a0.h, false, bf.h, (short)0, c[0][j], false, false);
            c[1][j] = __builtin_amdgcn_wmma_f32_16x16x32_f16(
                false, a1.h, false, bf.h, (short)0, c[1][j], false, false);
        }
    }

    // ---- Epilogue: out[b, u mod 256] += sum_rows w[m] * Y^2 ----
    if (tid < 128) smOut[tid] = 0.0f;
    __syncthreads();

    float wv0[8], wv1[8];
#pragma unroll
    for (int r = 0; r < 8; ++r) {
        wv0[r] = w[m0 + mw + sel * 8 + r];        // rows of C tile 0 in this lane
        wv1[r] = w[m0 + mw + 16 + sel * 8 + r];   // rows of C tile 1
    }

#pragma unroll
    for (int j = 0; j < 4; ++j) {
        float p = 0.0f;
#pragma unroll
        for (int r = 0; r < 8; ++r) {
            float y0 = c[0][j][r];
            float y1 = c[1][j][r];
            p += wv0[r] * y0 * y0 + wv1[r] * y1 * y1;
        }
        // combine half-wave row groups (M 0..7 with M 8..15)
        p += __shfl_xor(p, 16, 32);
        if (sel == 0)
            atomicAdd(&smOut[nw + 16 * j + l16], p);   // ds_add_f32
    }
    __syncthreads();

    // one global atomic per column per block (A^2+B^2 fold via mod 256)
    if (tid < 128) {
        const int b = m0 >> 11;                    // m0 / SEQ (tile within one b)
        const int u = (n0 + tid) & 255;
        atomicAdd(&out[b * UNITS + u], smOut[tid]);
    }
}

// ---------------------------------------------------------------------------
// Host-side launcher
// ---------------------------------------------------------------------------
extern "C" void kernel_launch(void* const* d_in, const int* in_sizes, int n_in,
                              void* d_out, int out_size, void* d_ws, size_t ws_size,
                              hipStream_t stream)
{
    const int*   d_indices = (const int*)  d_in[0];   // (B,S)
    const float* d_amp     = (const float*)d_in[1];   // (VOCAB,D)
    const float* d_phase   = (const float*)d_in[2];   // (VOCAB,D)
    const float* d_kernel  = (const float*)d_in[3];   // (UNITS,D,2)
    float*       out       = (float*)d_out;           // (B,UNITS)

    // Workspace layout
    char* ws = (char*)d_ws;
    _Float16* X    = (_Float16*)(ws);                                   // 65536*256 f16 = 32 MB
    _Float16* Wt   = (_Float16*)(ws + (size_t)M_TOK * KDIM * 2);        // 512*256 f16
    float*    wraw = (float*)   (ws + (size_t)M_TOK * KDIM * 2
                                    + (size_t)NDIM * KDIM * 2);         // 65536 f32
    float*    wsm  = (float*)   (ws + (size_t)M_TOK * KDIM * 2
                                    + (size_t)NDIM * KDIM * 2
                                    + (size_t)M_TOK * 4);               // 65536 f32

    // 1) pack rotation kernel into f16 weight matrix
    prep_w_kernel<<<NDIM, KDIM, 0, stream>>>(d_kernel, Wt);

    // 2) gather + normalize + phase rotate tokens into X, raw weights
    prep_tok_kernel<<<M_TOK, DIM, 0, stream>>>(d_indices, d_amp, d_phase, X, wraw);

    // 3) softmax over sequence per batch row
    softmax_kernel<<<BATCH, 256, 0, stream>>>(wraw, wsm);

    // 4) zero output accumulator
    zero_out_kernel<<<(BATCH * UNITS + 255) / 256, 256, 0, stream>>>(out, BATCH * UNITS);

    // 5) LDS-staged WMMA GEMM with fused weighted |.|^2 reduction
    dim3 grid(M_TOK / 128, NDIM / 128);
    gemm_reduce_kernel<<<grid, 256, 0, stream>>>(X, Wt, wsm, out);
}
